// RTM_2980707303681
// MI455X (gfx1250) — compile-verified, matrix operand order checked by reading, not run on hardware
//
#include <hip/hip_runtime.h>
#include <math.h>

typedef __attribute__((ext_vector_type(16))) __bf16 v16bf;
typedef __attribute__((ext_vector_type(8)))  __bf16 v8bf;
typedef __attribute__((ext_vector_type(8)))  float  v8f;

#define A_DIM 8
#define B_DIM 2048
#define C_DIM 512
#define TK_DIM 4
#define M_TOK (A_DIM * B_DIM)   // 16384
#define FF_DIM (4 * C_DIM)      // 2048

// ---------------------------------------------------------------------------
// Generic bf16 WMMA GEMM:  out[m,n] = sum_k A[m,k] * W[n,k]  (+ epilogue)
// A: M x K bf16 row-major, W: N x K bf16 row-major.
// Wave tile: 32(M) x 64(N)  -> 2 A-frags x 4 B-frags -> 8 WMMA / 12 b128 loads
// per k-step (vs 8/18 for a 16x128 tile): B fragments reused across two M
// sub-tiles, halving vmem pressure on the WMMA-bound inner loop.
// 8 waves/WG arranged 4(M) x 2(N) -> 128 x 128 block tile.
// EPI: 0 = + posbias[row%2048, col] -> f32
//      1 = plain -> f32
//      2 = + resid f32 -> f32
//      3 = + biasvec[col], exact GELU -> bf16
//      4 = + biasvec[col] -> f32
// ---------------------------------------------------------------------------
template <int EPI>
__global__ __launch_bounds__(256) void gemm_bf16_wmma(
    const __bf16* __restrict__ A, const __bf16* __restrict__ W,
    const float* __restrict__ resid, const float* __restrict__ bias,
    float* __restrict__ outF, __bf16* __restrict__ outB,
    int M, int N, int K) {
  const int lane  = threadIdx.x & 31;
  const int wave  = threadIdx.x >> 5;
  const int waveM = wave & 3;
  const int waveN = wave >> 2;
  const int m0 = blockIdx.x * 128 + waveM * 32;
  const int n0 = blockIdx.y * 128 + waveN * 64;
  const int r  = lane & 15;     // row within 16-row fragment
  const int hs = lane >> 4;     // K-half select

  // Fragment layout (16-bit A/B, 16x32): lane l<16 holds row l with
  // K in {0..7, 16..23}; lane l>=16 holds row l-16 with K in {8..15, 24..31}.
  const __bf16* Ar0 = A + (size_t)(m0 + r) * K + hs * 8;
  const __bf16* Ar1 = Ar0 + (size_t)16 * K;
  const __bf16* Wr0 = W + (size_t)(n0 + r) * K + hs * 8;

  v8f acc[2][4] = {};

  for (int k0 = 0; k0 < K; k0 += 32) {
    union { v16bf v; v8bf h[2]; } a0, a1;
    a0.h[0] = *(const v8bf*)(Ar0 + k0);
    a0.h[1] = *(const v8bf*)(Ar0 + k0 + 16);
    a1.h[0] = *(const v8bf*)(Ar1 + k0);
    a1.h[1] = *(const v8bf*)(Ar1 + k0 + 16);
    if (k0 + 32 < K) {
      __builtin_prefetch(Ar0 + k0 + 32, 0, 1);
      __builtin_prefetch(Ar1 + k0 + 32, 0, 1);
    }

#pragma unroll
    for (int ni = 0; ni < 4; ++ni) {
      const __bf16* Wrow = Wr0 + (size_t)(ni * 16) * K + k0;
      union { v16bf v; v8bf h[2]; } bf;
      bf.h[0] = *(const v8bf*)(Wrow);
      bf.h[1] = *(const v8bf*)(Wrow + 16);
      acc[0][ni] = __builtin_amdgcn_wmma_f32_16x16x32_bf16(
          false, a0.v, false, bf.v, (short)0, acc[0][ni], false, false);
      acc[1][ni] = __builtin_amdgcn_wmma_f32_16x16x32_bf16(
          false, a1.v, false, bf.v, (short)0, acc[1][ni], false, false);
    }
  }

  // C/D layout: lane l<16 -> col = n-tile base + l, elem e -> row base + e;
  //             lane l>=16 -> same cols, rows base + 8 + e.
#pragma unroll
  for (int mi = 0; mi < 2; ++mi) {
#pragma unroll
    for (int ni = 0; ni < 4; ++ni) {
      const int col = n0 + ni * 16 + r;
#pragma unroll
      for (int e = 0; e < 8; ++e) {
        const int row = m0 + mi * 16 + hs * 8 + e;
        float v = acc[mi][ni][e];
        const size_t oidx = (size_t)row * N + col;
        if constexpr (EPI == 0) {
          outF[oidx] = v + bias[(size_t)(row & (B_DIM - 1)) * N + col];
        } else if constexpr (EPI == 1) {
          outF[oidx] = v;
        } else if constexpr (EPI == 2) {
          outF[oidx] = v + resid[oidx];
        } else if constexpr (EPI == 3) {
          v += bias[col];
          v = 0.5f * v * (1.0f + erff(v * 0.70710678118654752f));
          outB[oidx] = (__bf16)v;
        } else {
          outF[oidx] = v + bias[col];
        }
      }
    }
  }
}

// ---------------------------------------------------------------------------
// LayerNorm over C=512, one block (256 threads) per token.
// ln1: act_bf16 = LN(s)                 (input to V projection)
// ln2: s = LN(s)+s (f32), act_bf16 = s  (input to fc1)
// ---------------------------------------------------------------------------
__global__ __launch_bounds__(256) void ln1_kernel(
    const float* __restrict__ S, const float* __restrict__ g,
    const float* __restrict__ b, __bf16* __restrict__ act) {
  __shared__ float r1[256], r2[256];
  const int t = threadIdx.x;
  const size_t base = (size_t)blockIdx.x * C_DIM;
  float x0 = S[base + t], x1 = S[base + t + 256];
  r1[t] = x0 + x1;
  r2[t] = x0 * x0 + x1 * x1;
  __syncthreads();
  for (int o = 128; o > 0; o >>= 1) {
    if (t < o) { r1[t] += r1[t + o]; r2[t] += r2[t + o]; }
    __syncthreads();
  }
  const float mean = r1[0] * (1.0f / C_DIM);
  const float var  = r2[0] * (1.0f / C_DIM) - mean * mean;
  const float rstd = rsqrtf(var + 1e-5f);
  act[base + t]       = (__bf16)((x0 - mean) * rstd * g[t] + b[t]);
  act[base + t + 256] = (__bf16)((x1 - mean) * rstd * g[t + 256] + b[t + 256]);
}

__global__ __launch_bounds__(256) void ln2_kernel(
    float* __restrict__ S, const float* __restrict__ g,
    const float* __restrict__ b, __bf16* __restrict__ act) {
  __shared__ float r1[256], r2[256];
  const int t = threadIdx.x;
  const size_t base = (size_t)blockIdx.x * C_DIM;
  float x0 = S[base + t], x1 = S[base + t + 256];
  r1[t] = x0 + x1;
  r2[t] = x0 * x0 + x1 * x1;
  __syncthreads();
  for (int o = 128; o > 0; o >>= 1) {
    if (t < o) { r1[t] += r1[t + o]; r2[t] += r2[t + o]; }
    __syncthreads();
  }
  const float mean = r1[0] * (1.0f / C_DIM);
  const float var  = r2[0] * (1.0f / C_DIM) - mean * mean;
  const float rstd = rsqrtf(var + 1e-5f);
  float y0 = (x0 - mean) * rstd * g[t] + b[t] + x0;
  float y1 = (x1 - mean) * rstd * g[t + 256] + b[t + 256] + x1;
  S[base + t] = y0;           act[base + t] = (__bf16)y0;
  S[base + t + 256] = y1;     act[base + t + 256] = (__bf16)y1;
}

// ---------------------------------------------------------------------------
// Causal prefix sum over the sequence axis (w == 1 exactly: softmax of a
// singleton). One block per batch a, thread = channel c, fully coalesced.
// V f32 -> IMV bf16 (f32 running accumulator).
// ---------------------------------------------------------------------------
__global__ __launch_bounds__(512) void cumsum_kernel(
    const float* __restrict__ V, __bf16* __restrict__ IMV) {
  const size_t base = (size_t)blockIdx.x * B_DIM * C_DIM + threadIdx.x;
  float acc = 0.0f;
  for (int j = 0; j < B_DIM; ++j) {
    acc += V[base + (size_t)j * C_DIM];
    IMV[base + (size_t)j * C_DIM] = (__bf16)acc;
  }
}

// ---------------------------------------------------------------------------
// Positional bias:  bias[j,2t] = sin(j / 10000^(t/128)),  bias[j,2t+1] = cos.
// ---------------------------------------------------------------------------
__global__ __launch_bounds__(256) void posbias_kernel(float* __restrict__ bias) {
  const int idx = blockIdx.x * 256 + threadIdx.x;  // 2048 * 256 threads
  const int t = idx & 255;
  const int j = idx >> 8;
  const float ang = (float)j * powf(10000.0f, -(float)t * (1.0f / 128.0f));
  bias[(size_t)j * C_DIM + 2 * t]     = sinf(ang);
  bias[(size_t)j * C_DIM + 2 * t + 1] = cosf(ang);
}

__global__ __launch_bounds__(256) void f32_to_bf16(
    const float* __restrict__ in, __bf16* __restrict__ out, int n) {
  const int i = blockIdx.x * 256 + threadIdx.x;
  if (i < n) out[i] = (__bf16)in[i];
}

// ---------------------------------------------------------------------------
extern "C" void kernel_launch(void* const* d_in, const int* in_sizes, int n_in,
                              void* d_out, int out_size, void* d_ws, size_t ws_size,
                              hipStream_t stream) {
  const float* x      = (const float*)d_in[0];
  const float* weight = (const float*)d_in[1];
  // d_in[2] = Wq, d_in[3] = Wk : unused (softmax over singleton == 1.0)
  const float* Wv     = (const float*)d_in[4];
  const float* Wo     = (const float*)d_in[5];
  const float* ln1_g  = (const float*)d_in[6];
  const float* ln1_b  = (const float*)d_in[7];
  const float* ln2_g  = (const float*)d_in[8];
  const float* ln2_b  = (const float*)d_in[9];
  const float* fc1_w  = (const float*)d_in[10];
  const float* fc1_b  = (const float*)d_in[11];
  const float* fc2_w  = (const float*)d_in[12];
  const float* fc2_b  = (const float*)d_in[13];
  float* out = (float*)d_out;

  // ---- workspace layout (256B aligned) ----
  char* ws = (char*)d_ws;
  size_t off = 0;
  auto take = [&](size_t bytes) { char* p = ws + off; off = (off + bytes + 255) & ~(size_t)255; return p; };
  float*  S    = (float*)take((size_t)M_TOK * C_DIM * 4);          // 32 MB running state
  float*  BIAS = (float*)take((size_t)B_DIM * C_DIM * 4);          //  4 MB pos bias
  __bf16* ACT  = (__bf16*)take((size_t)M_TOK * C_DIM * 2);         // 16 MB bf16 activations
  __bf16* IMV  = (__bf16*)take((size_t)M_TOK * C_DIM * 2);         // 16 MB cumsum(v)
  char*   BIG  = take((size_t)M_TOK * FF_DIM * 2);                 // 64 MB (V f32 | hid bf16)
  __bf16* Wb   = (__bf16*)take((size_t)C_DIM * C_DIM * 2);
  __bf16* WVb  = (__bf16*)take((size_t)TK_DIM * C_DIM * C_DIM * 2);
  __bf16* WOb  = (__bf16*)take((size_t)TK_DIM * C_DIM * C_DIM * 2);
  __bf16* F1b  = (__bf16*)take((size_t)FF_DIM * C_DIM * 2);
  __bf16* F2b  = (__bf16*)take((size_t)C_DIM * FF_DIM * 2);
  float*  Vf32 = (float*)BIG;
  __bf16* HID  = (__bf16*)BIG;

  auto cvt = [&](const float* src, __bf16* dst, int n) {
    f32_to_bf16<<<(n + 255) / 256, 256, 0, stream>>>(src, dst, n);
  };

  // ---- one-time conversions + positional bias ----
  cvt(weight, Wb, C_DIM * C_DIM);
  cvt(Wv, WVb, TK_DIM * C_DIM * C_DIM);
  cvt(Wo, WOb, TK_DIM * C_DIM * C_DIM);
  cvt(fc1_w, F1b, FF_DIM * C_DIM);
  cvt(fc2_w, F2b, C_DIM * FF_DIM);
  cvt(x, ACT, M_TOK * C_DIM);
  posbias_kernel<<<B_DIM, 256, 0, stream>>>(BIAS);

  const dim3 blk(256);
  const dim3 gC(M_TOK / 128, C_DIM / 128);   // N = 512
  const dim3 gF(M_TOK / 128, FF_DIM / 128);  // N = 2048

  // s = x @ weight^T + posbias
  gemm_bf16_wmma<0><<<gC, blk, 0, stream>>>(ACT, Wb, nullptr, BIAS, S, nullptr,
                                            M_TOK, C_DIM, C_DIM);

  for (int a = 0; a < TK_DIM; ++a) {
    // h = LN1(s)  (bf16)
    ln1_kernel<<<M_TOK, blk, 0, stream>>>(S, ln1_g, ln1_b, ACT);
    // v = h @ Wv[a]^T  (f32)
    gemm_bf16_wmma<1><<<gC, blk, 0, stream>>>(ACT, WVb + (size_t)a * C_DIM * C_DIM,
                                              nullptr, nullptr, Vf32, nullptr,
                                              M_TOK, C_DIM, C_DIM);
    // imv = cumsum(v) over seq  (bf16)
    cumsum_kernel<<<A_DIM, 512, 0, stream>>>(Vf32, IMV);
    // s = s + imv @ Wo[a]^T
    gemm_bf16_wmma<2><<<gC, blk, 0, stream>>>(IMV, WOb + (size_t)a * C_DIM * C_DIM,
                                              S, nullptr, S, nullptr,
                                              M_TOK, C_DIM, C_DIM);
    // s = LN2(s) + s  (f32 + bf16 copy)
    ln2_kernel<<<M_TOK, blk, 0, stream>>>(S, ln2_g, ln2_b, ACT);
    // hid = gelu(s @ fc1^T + b1)  (bf16)
    gemm_bf16_wmma<3><<<gF, blk, 0, stream>>>(ACT, F1b, nullptr, fc1_b,
                                              nullptr, HID, M_TOK, FF_DIM, C_DIM);
    // s = hid @ fc2^T + b2  (f32; last iteration -> d_out)
    float* dst = (a == TK_DIM - 1) ? out : S;
    gemm_bf16_wmma<4><<<gC, blk, 0, stream>>>(HID, F2b, nullptr, fc2_b,
                                              dst, nullptr, M_TOK, C_DIM, FF_DIM);
  }
  (void)in_sizes; (void)n_in; (void)out_size; (void)ws_size;
}
